// GCN_28905129902086
// MI455X (gfx1250) — matrix operand order repaired and checked by reading, not compiled, not run on hardware
//
#include <hip/hip_runtime.h>
#include <hip/hip_bf16.h>

// CDNA5 / gfx1250 GCN forward. Heavy GEMMs on v_wmma_f32_16x16x32_f16.
// Fragment layouts per CDNA5 ISA 7.12.2:
//   A (16x32 f16): lane half h=lane>>4, m=lane&15; VGPR v holds K pair
//       k = (v>>2)*16 + h*8 + (v&3)*2 + {0,1}
//   B (32x16 f16): n=lane&15; VGPR v holds K pair k = (v>>2)*16 + h*8 + (v&3)*2 + {0,1}, same as A
//   C/D (16x16 f32): reg r, half h -> (M = r + 8h, N = lane&15)

typedef __attribute__((ext_vector_type(16))) _Float16 v16h;
typedef __attribute__((ext_vector_type(8)))  float    v8f;

#define N_NODES    100000
#define N_EDGES    1250000
#define N_FEAT     128
#define NHID       64
#define NUM_GRAPHS 256

__device__ __forceinline__ unsigned pack2(float lo, float hi) {
  union { _Float16 h[2]; unsigned u; } p;
  p.h[0] = (_Float16)lo; p.h[1] = (_Float16)hi;
  return p.u;
}

__device__ __forceinline__ v16h load_frag8(const unsigned* __restrict__ p) {
  union { unsigned u[8]; v16h v; } f;
#pragma unroll
  for (int i = 0; i < 8; ++i) f.u[i] = p[i];
  return f.v;
}

__device__ __forceinline__ v8f wmma16(v16h a, v16h b, v8f c) {
  return __builtin_amdgcn_wmma_f32_16x16x32_f16(false, a, false, b,
                                                (short)0, c, false, false);
}

// ---------------------------------------------------------------- init
__global__ void k_init(float* __restrict__ deg, float* __restrict__ pooled) {
  int t = blockIdx.x * blockDim.x + threadIdx.x;
  if (t < N_NODES) deg[t] = 1.0f;                 // self-loop weight
  if (t < NUM_GRAPHS * NHID) pooled[t] = 0.0f;
}

// ------------------------------------------------- edge MLP -> edge_w, deg
// 16 edges per wave, 8 waves (128 edges) per block.
__global__ __launch_bounds__(256) void k_edge_mlp(
    const float* __restrict__ edge_attr,
    const float* __restrict__ We1, const float* __restrict__ be1,
    const float* __restrict__ We2, const float* __restrict__ be2,
    const float* __restrict__ We3, const float* __restrict__ be3,
    const int* __restrict__ eidx,
    float* __restrict__ edge_w, float* __restrict__ deg)
{
  __shared__ float sWe1[5 * 64];
  __shared__ float sbe1[64], sbe2[64], sWe3[64];
  __shared__ unsigned sB[4][2][32][8];  // [ntile][kstep][lane][vgpr], 8 KB

  const int tid = threadIdx.x;
  for (int i = tid; i < 5 * 64; i += 256) sWe1[i] = We1[i];
  if (tid < 64) { sbe1[tid] = be1[tid]; sbe2[tid] = be2[tid]; sWe3[tid] = We3[tid]; }
  // Pre-pack We2 [64x64] into B-fragment layout (f16 pairs).
  for (int i = tid; i < 4 * 2 * 32 * 8; i += 256) {
    int v = i & 7, l = (i >> 3) & 31, s = (i >> 8) & 1, t = (i >> 9) & 3;
    int k = s * 32 + (v >> 2) * 16 + (l >> 4) * 8 + (v & 3) * 2;
    int n = t * 16 + (l & 15);
    sB[t][s][l][v] = pack2(We2[k * 64 + n], We2[(k + 1) * 64 + n]);
  }
  __syncthreads();

  const int lane  = tid & 31;
  const int wave  = tid >> 5;
  const int h     = lane >> 4;
  const int n     = lane & 15;
  const int ebase = blockIdx.x * 128 + wave * 16;

  // both halves fetch the edge_attr row for m = n
  const int erow = ebase + n;
  const int ec   = erow < N_EDGES ? erow : (N_EDGES - 1);
  if (erow + 128 < N_EDGES) __builtin_prefetch(&edge_attr[(erow + 128) * 5], 0, 3);
  float ea[5];
#pragma unroll
  for (int j = 0; j < 5; ++j) ea[j] = edge_attr[ec * 5 + j];

  // h1 = relu(ea @ We1 + be1) built straight into A-fragments (K = 64 -> 2 steps)
  v16h A[2];
#pragma unroll
  for (int s = 0; s < 2; ++s) {
#pragma unroll
    for (int v = 0; v < 8; ++v) {
#pragma unroll
      for (int d = 0; d < 2; ++d) {
        int k = s * 32 + (v >> 2) * 16 + h * 8 + (v & 3) * 2 + d;
        float a = sbe1[k];
#pragma unroll
        for (int j = 0; j < 5; ++j) a += ea[j] * sWe1[j * 64 + k];
        A[s][2 * v + d] = (_Float16)(a > 0.f ? a : 0.f);
      }
    }
  }

  // h2 = h1 @ We2 : 4 N-tiles x 2 K-steps = 8 WMMA
  v8f acc[4];
#pragma unroll
  for (int t = 0; t < 4; ++t) {
    v8f c = {};
#pragma unroll
    for (int s = 0; s < 2; ++s)
      c = wmma16(A[s], load_frag8(&sB[t][s][lane][0]), c);
    acc[t] = c;
  }

  // edge_w = relu(relu(h2 + be2) @ We3 + be3); reduce over N within each half
  float we3v[4], be2v[4];
#pragma unroll
  for (int t = 0; t < 4; ++t) { we3v[t] = sWe3[t * 16 + n]; be2v[t] = sbe2[t * 16 + n]; }

  float myw = 0.f;
#pragma unroll
  for (int r = 0; r < 8; ++r) {
    float p = 0.f;
#pragma unroll
    for (int t = 0; t < 4; ++t) {
      float hv = acc[t][r] + be2v[t];
      p += (hv > 0.f ? hv : 0.f) * we3v[t];
    }
    p += __shfl_xor(p, 1);
    p += __shfl_xor(p, 2);
    p += __shfl_xor(p, 4);
    p += __shfl_xor(p, 8);
    if (n == r) myw = p;          // row (r + 8h) owner
  }

  if (n < 8) {
    int m = ebase + n + 8 * h;
    if (m < N_EDGES) {
      float w = myw + be3[0];
      w = w > 0.f ? w : 0.f;
      edge_w[m] = w;
      if (w != 0.f) atomicAdd(&deg[eidx[N_EDGES + m]], w);
    }
  }
}

// ---------------------------------------------------------------- dinv
__global__ void k_dinv(const float* __restrict__ deg, float* __restrict__ dinv) {
  int t = blockIdx.x * blockDim.x + threadIdx.x;
  if (t < N_NODES) {
    float d = deg[t];
    dinv[t] = d > 0.f ? rsqrtf(d) : 0.f;
  }
}

// --------------------------------- xw = x @ Wg ; agg init = xw * dinv^2
__global__ __launch_bounds__(256) void k_xw(
    const float* __restrict__ x, const float* __restrict__ Wg,
    const float* __restrict__ dinv,
    float* __restrict__ xw, float* __restrict__ agg)
{
  __shared__ unsigned sB[4][4][32][8];  // [ntile][kstep][lane][vgpr], 16 KB
  const int tid = threadIdx.x;
  for (int i = tid; i < 4 * 4 * 32 * 8; i += 256) {
    int v = i & 7, l = (i >> 3) & 31, s = (i >> 8) & 3, t = (i >> 10) & 3;
    int k = s * 32 + (v >> 2) * 16 + (l >> 4) * 8 + (v & 3) * 2;
    int n = t * 16 + (l & 15);
    sB[t][s][l][v] = pack2(Wg[k * 64 + n], Wg[(k + 1) * 64 + n]);
  }
  __syncthreads();

  const int lane  = tid & 31, wave = tid >> 5;
  const int h     = lane >> 4, n = lane & 15;
  const int nbase = blockIdx.x * 128 + wave * 16;
  const int mrow  = nbase + n;
  const int mc    = mrow < N_NODES ? mrow : (N_NODES - 1);

  v8f acc[4] = {};
#pragma unroll
  for (int s = 0; s < 4; ++s) {            // K = 128 -> 4 steps
    v16h A;
#pragma unroll
    for (int v = 0; v < 8; ++v) {
      int k = s * 32 + (v >> 2) * 16 + h * 8 + (v & 3) * 2;
      float2 xv = *(const float2*)&x[mc * N_FEAT + k];
      A[2 * v]     = (_Float16)xv.x;
      A[2 * v + 1] = (_Float16)xv.y;
    }
#pragma unroll
    for (int t = 0; t < 4; ++t)
      acc[t] = wmma16(A, load_frag8(&sB[t][s][lane][0]), acc[t]);
  }

#pragma unroll
  for (int r = 0; r < 8; ++r) {
    int m = nbase + r + 8 * h;
    if (m < N_NODES) {
      float di = dinv[m];
      float di2 = di * di;
#pragma unroll
      for (int t = 0; t < 4; ++t) {
        float val = acc[t][r];
        xw [m * 64 + t * 16 + n] = val;
        agg[m * 64 + t * 16 + n] = val * di2;   // self-loop term
      }
    }
  }
}

// ------------------------------------------- scatter: agg += norm * xw[row]
__global__ void k_aggregate(const int* __restrict__ eidx,
                            const float* __restrict__ edge_w,
                            const float* __restrict__ dinv,
                            const float* __restrict__ xw,
                            float* __restrict__ agg)
{
  long long t = (long long)blockIdx.x * blockDim.x + threadIdx.x;
  int e = (int)(t >> 4);
  int c = (int)(t & 15);
  if (e >= N_EDGES) return;
  float w = edge_w[e];
  if (w == 0.f) return;                         // ReLU-killed edge: no traffic
  int row = eidx[e], col = eidx[N_EDGES + e];
  float norm = dinv[row] * w * dinv[col];
  float4 v = *(const float4*)&xw[row * 64 + 4 * c];
  float* dst = &agg[col * 64 + 4 * c];
  atomicAdd(dst + 0, v.x * norm);
  atomicAdd(dst + 1, v.y * norm);
  atomicAdd(dst + 2, v.z * norm);
  atomicAdd(dst + 3, v.w * norm);
}

// ------------------------------- node = relu(agg + b_g); pooled[batch] += node
__global__ void k_pool(const float* __restrict__ agg, const float* __restrict__ bg,
                       const int* __restrict__ batch, float* __restrict__ pooled)
{
  int t = blockIdx.x * blockDim.x + threadIdx.x;
  int nn = t >> 4, c = t & 15;
  if (nn >= N_NODES) return;
  int g = batch[nn];
  float4 v = *(const float4*)&agg[nn * 64 + 4 * c];
  float4 b = *(const float4*)&bg[4 * c];
  float* dst = &pooled[g * 64 + 4 * c];
  atomicAdd(dst + 0, fmaxf(v.x + b.x, 0.f));
  atomicAdd(dst + 1, fmaxf(v.y + b.y, 0.f));
  atomicAdd(dst + 2, fmaxf(v.z + b.z, 0.f));
  atomicAdd(dst + 3, fmaxf(v.w + b.w, 0.f));
}

// ------------------- logits = relu(pooled @ Wb1 + bb1) @ Wb2 + bb2 (16 rows/blk)
__global__ __launch_bounds__(32) void k_graph_mlp(
    const float* __restrict__ pooled,
    const float* __restrict__ Wb1, const float* __restrict__ bb1,
    const float* __restrict__ Wb2, const float* __restrict__ bb2,
    float* __restrict__ out)
{
  __shared__ unsigned sB[4][2][32][8];
  __shared__ float sbb1[64], sWb2[64];
  const int lane = threadIdx.x & 31;
  for (int i = lane; i < 4 * 2 * 32 * 8; i += 32) {
    int v = i & 7, l = (i >> 3) & 31, s = (i >> 8) & 1, t = (i >> 9) & 3;
    int k = s * 32 + (v >> 2) * 16 + (l >> 4) * 8 + (v & 3) * 2;
    int n = t * 16 + (l & 15);
    sB[t][s][l][v] = pack2(Wb1[k * 64 + n], Wb1[(k + 1) * 64 + n]);
  }
  sbb1[lane] = bb1[lane];   sbb1[lane + 32] = bb1[lane + 32];
  sWb2[lane] = Wb2[lane];   sWb2[lane + 32] = Wb2[lane + 32];
  __syncthreads();

  const int h = lane >> 4, n = lane & 15;
  const int m = blockIdx.x * 16 + n;

  v16h A[2];
#pragma unroll
  for (int s = 0; s < 2; ++s)
#pragma unroll
    for (int v = 0; v < 8; ++v) {
      int k = s * 32 + (v >> 2) * 16 + h * 8 + (v & 3) * 2;
      A[s][2 * v]     = (_Float16)pooled[m * 64 + k];
      A[s][2 * v + 1] = (_Float16)pooled[m * 64 + k + 1];
    }

  v8f acc[4];
#pragma unroll
  for (int t = 0; t < 4; ++t) {
    v8f c = {};
#pragma unroll
    for (int s = 0; s < 2; ++s)
      c = wmma16(A[s], load_frag8(&sB[t][s][lane][0]), c);
    acc[t] = c;
  }

  float wbv[4], bbv[4];
#pragma unroll
  for (int t = 0; t < 4; ++t) { wbv[t] = sWb2[t * 16 + n]; bbv[t] = sbb1[t * 16 + n]; }

  float myv = 0.f;
#pragma unroll
  for (int r = 0; r < 8; ++r) {
    float p = 0.f;
#pragma unroll
    for (int t = 0; t < 4; ++t) {
      float z = acc[t][r] + bbv[t];
      p += (z > 0.f ? z : 0.f) * wbv[t];
    }
    p += __shfl_xor(p, 1);
    p += __shfl_xor(p, 2);
    p += __shfl_xor(p, 4);
    p += __shfl_xor(p, 8);
    if (n == r) myv = p;
  }

  if (n < 8) out[blockIdx.x * 16 + n + 8 * h] = myv + bb2[0];
}

// ---------------------------------------------------------------- launch
extern "C" void kernel_launch(void* const* d_in, const int* in_sizes, int n_in,
                              void* d_out, int out_size, void* d_ws, size_t ws_size,
                              hipStream_t stream) {
  const float* x    = (const float*)d_in[0];
  const float* eatt = (const float*)d_in[1];
  const float* We1  = (const float*)d_in[2];
  const float* be1  = (const float*)d_in[3];
  const float* We2  = (const float*)d_in[4];
  const float* be2  = (const float*)d_in[5];
  const float* We3  = (const float*)d_in[6];
  const float* be3  = (const float*)d_in[7];
  const float* Wg   = (const float*)d_in[8];
  const float* bg   = (const float*)d_in[9];
  const float* Wb1  = (const float*)d_in[10];
  const float* bb1  = (const float*)d_in[11];
  const float* Wb2  = (const float*)d_in[12];
  const float* bb2  = (const float*)d_in[13];
  const int*   eidx = (const int*)d_in[14];
  const int*   batch= (const int*)d_in[15];
  float* out = (float*)d_out;

  // workspace carve-up (floats)
  float* ws     = (float*)d_ws;
  float* edge_w = ws;                                   // E
  float* deg    = edge_w + N_EDGES;                     // N
  float* dinv   = deg + N_NODES;                        // N
  float* xw     = dinv + N_NODES;                       // N*64
  float* agg    = xw + (size_t)N_NODES * NHID;          // N*64
  float* pooled = agg + (size_t)N_NODES * NHID;         // 256*64

  k_init<<<(N_NODES + 255) / 256, 256, 0, stream>>>(deg, pooled);
  k_edge_mlp<<<(N_EDGES + 127) / 128, 256, 0, stream>>>(
      eatt, We1, be1, We2, be2, We3, be3, eidx, edge_w, deg);
  k_dinv<<<(N_NODES + 255) / 256, 256, 0, stream>>>(deg, dinv);
  k_xw<<<(N_NODES + 127) / 128, 256, 0, stream>>>(x, Wg, dinv, xw, agg);
  k_aggregate<<<(N_EDGES * 16) / 256, 256, 0, stream>>>(eidx, edge_w, dinv, xw, agg);
  k_pool<<<(N_NODES * 16) / 256, 256, 0, stream>>>(agg, bg, batch, pooled);
  k_graph_mlp<<<NUM_GRAPHS / 16, 32, 0, stream>>>(pooled, Wb1, bb1, Wb2, bb2, out);
  (void)in_sizes; (void)n_in; (void)out_size; (void)ws_size;
}
